// RegionProposalNetwork_27152783246007
// MI455X (gfx1250) — matrix operand order repaired and checked by reading, not compiled, not run on hardware
//
#include <hip/hip_runtime.h>
#include <hip/hip_bf16.h>
#include <math.h>

typedef __attribute__((ext_vector_type(16))) __bf16 v16bf;
typedef __attribute__((ext_vector_type(8)))  float  v8f;
typedef unsigned int u32x4_t __attribute__((ext_vector_type(4)));
typedef int          i32x4_t __attribute__((ext_vector_type(4)));
typedef int          i32x8_t __attribute__((ext_vector_type(8)));
typedef __attribute__((address_space(3))) unsigned short lds_us_t;

union FragBF { uint4 q[2]; v16bf v; };

#if __has_builtin(__builtin_amdgcn_tensor_load_to_lds)
#define USE_TDM 1
#else
#define USE_TDM 0
#endif

__device__ __forceinline__ void wait_tensorcnt0() {
#if __has_builtin(__builtin_amdgcn_s_wait_tensorcnt)
  __builtin_amdgcn_s_wait_tensorcnt(0);
#else
  asm volatile("s_wait_tensorcnt 0x0" ::: "memory");
#endif
}

#define N_IMG   4
#define NPOS    4096              /* 64*64 */
#define MTOT    16384             /* 4*4096 */
#define KTOT    4608              /* 512*9 */
#define NA_TOT  36864             /* 4096*9 */
#define PRE_NMS 3000
#define POST_NMS 300
#define CAND_MAX 4096

/* d_out layout (floats, concatenated tuple) */
#define OFF_LOCS   0
#define OFF_SCORES 589824
#define OFF_ROIS   884736
#define OFF_VALID  889536
#define OFF_ANCH   890736

/* workspace layout (bytes) */
static const size_t WS_XB   = 0;               /* 16384*512 bf16  = 16777216 */
static const size_t WS_WB   = 16777216;        /*  512*4608 bf16  =  4718592 */
static const size_t WS_WH   = 21495808;        /*   64*512  bf16  =    65536 */
static const size_t WS_HB   = 21561344;        /* 16384*512 bf16  = 16777216 */
static const size_t WS_SC   = 38338560;        /* 4*36864 f32     =   589824 */
static const size_t WS_BX   = 38928384;        /* 4*36864*4 f32   =  2359296 */
static const size_t WS_HIST = 41287680;        /* 4*256 i32       =     4096 */
static const size_t WS_TOPB = 41291776;        /* 4*3000*4 f32    =   192000 */
static const size_t WS_TOPS = 41483776;        /* 4*3000 f32      =    48000 */

__device__ __forceinline__ unsigned short f2bf(float f) {
  unsigned u = __float_as_uint(f);
  u += 0x7FFFu + ((u >> 16) & 1u);
  return (unsigned short)(u >> 16);
}

#if USE_TDM
/* TDM: async 2D tile load Wb[row0..row0+128)[k0..k0+32) -> LDS (padded rows
   of 40 shorts: pad_interval=3 -> pad every 16 DWORDs, pad_amount=3 -> 4 DWORDs). */
__device__ __forceinline__ void tdm_load_B(const unsigned short* gptr, unsigned lds_off) {
  unsigned long long ga = (unsigned long long)(size_t)gptr;
  u32x4_t g0;
  g0[0] = 1u;                                       /* count=1, user mode      */
  g0[1] = lds_off;                                  /* lds_addr (bytes)        */
  g0[2] = (unsigned)(ga & 0xFFFFFFFFu);             /* global_addr[31:0]       */
  g0[3] = (unsigned)((ga >> 32) & 0x01FFFFFFu) | (2u << 30); /* addr[56:32], type=2 */
  i32x8_t g1;
  g1[0] = (int)((1u << 16)      /* data_size = 2 bytes                */
        |       (1u << 20)      /* pad_enable                         */
        |       (3u << 22)      /* pad_interval: every 16 DWORDs      */
        |       (3u << 25));    /* pad_amount: 4 DWORDs (16 B)        */
  g1[1] = (int)(((unsigned)KTOT & 0xFFFFu) << 16);  /* tensor_dim0[15:0] @79:48 */
  g1[2] = (int)((((unsigned)KTOT >> 16) & 0xFFFFu) | (512u << 16)); /* dim0 hi, dim1 lo */
  g1[3] = (int)(0u | (32u << 16));                  /* dim1 hi, tile_dim0=32    */
  g1[4] = (int)(128u);                              /* tile_dim1=128, tile_dim2=0 */
  g1[5] = (int)(KTOT);                              /* tensor_dim0_stride[31:0] */
  g1[6] = 0;                                        /* stride0 hi, stride1 lo   */
  g1[7] = 0;
  i32x4_t z4 = {0, 0, 0, 0};
  i32x8_t z8 = {0, 0, 0, 0, 0, 0, 0, 0};
  __builtin_amdgcn_tensor_load_to_lds(g0, g1, z4, z4, z8, 0);
}
#endif

/* ---------------- precision / layout transforms ---------------- */

extern "C" __global__ void k_cvt_x(const float* __restrict__ x,
                                   unsigned short* __restrict__ xb) {
  int gid = blockIdx.x * blockDim.x + threadIdx.x;
  if (gid >= N_IMG * 512 * NPOS) return;
  int n   = gid / (512 * NPOS);
  int rem = gid % (512 * NPOS);
  int c   = rem / NPOS;
  int pos = rem % NPOS;
  xb[((size_t)n * NPOS + pos) * 512 + c] = f2bf(x[gid]);
}

extern "C" __global__ void k_cvt_w(const float* __restrict__ W1,
                                   unsigned short* __restrict__ Wb) {
  int gid = blockIdx.x * blockDim.x + threadIdx.x;
  if (gid >= 512 * KTOT) return;
  int o = gid / KTOT;
  int k = gid % KTOT;
  int c = k / 9;
  int r = k % 9;             /* tap index kh*3+kw */
  Wb[(size_t)o * KTOT + r * 512 + c] = f2bf(W1[gid]);
}

extern "C" __global__ void k_cvt_hw(const float* __restrict__ Ws,
                                    const float* __restrict__ Wl,
                                    unsigned short* __restrict__ Wh) {
  int gid = blockIdx.x * blockDim.x + threadIdx.x;
  if (gid >= 64 * 512) return;
  int ch = gid >> 9;
  int c  = gid & 511;
  float v = (ch < 18) ? Ws[ch * 512 + c]
          : (ch < 54) ? Wl[(ch - 18) * 512 + c] : 0.f;
  Wh[(size_t)ch * 512 + c] = f2bf(v);
}

extern "C" __global__ void k_zero_hist(int* __restrict__ hist) {
  int gid = blockIdx.x * blockDim.x + threadIdx.x;
  if (gid < N_IMG * 256) hist[gid] = 0;
}

/* ---------------- 3x3 conv as implicit GEMM (WMMA bf16) ----------------
   A tile: manual loads (halo + zero-fill); B tile: TDM async DMA when available. */

extern "C" __global__ __launch_bounds__(256) void k_conv_wmma(
    const unsigned short* __restrict__ xb,   /* [16384][512] bf16 NHWC   */
    const unsigned short* __restrict__ Wb,   /* [512][4608]  bf16        */
    const float*          __restrict__ b1,
    unsigned short*       __restrict__ hb)   /* [16384][512] bf16 NHWC   */
{
  __shared__ unsigned short As[2][128][40];
  __shared__ unsigned short Bs[2][128][40];

  const int tid  = threadIdx.x;
  const int lane = tid & 31;
  const int wave = tid >> 5;
  const int m0   = blockIdx.x * 128;
  const int n0   = blockIdx.y * 128;

  /* loader: thread -> (row, half-of-32-K) */
  const int lrow  = tid >> 1;
  const int lhalf = tid & 1;
  const int mg    = m0 + lrow;
  const int img   = mg >> 12;
  const int pos   = mg & 4095;
  const int py    = pos >> 6;
  const int px    = pos & 63;
  const int nrow  = n0 + lrow;
  (void)nrow;

  v8f acc[8];
  v8f vz = {};
#pragma unroll
  for (int t = 0; t < 8; ++t) acc[t] = vz;

  uint4 ra0, ra1;
#if !USE_TDM
  uint4 rb0, rb1;
#endif

#define CONV_LOADA(STEP)                                                      \
  {                                                                           \
    int r_  = (STEP) >> 4;                                                    \
    int c0_ = ((STEP) & 15) * 32 + lhalf * 16;                                \
    int yy_ = py + (r_ / 3) - 1;                                              \
    int xs_ = px + (r_ % 3) - 1;                                              \
    if ((unsigned)yy_ < 64u && (unsigned)xs_ < 64u) {                         \
      const uint4* p_ = (const uint4*)(xb +                                   \
          (((size_t)img * NPOS + yy_ * 64 + xs_) * 512 + c0_));               \
      ra0 = p_[0]; ra1 = p_[1];                                               \
      __builtin_prefetch((const char*)p_ + 1024, 0, 1);                       \
    } else {                                                                  \
      ra0.x = ra0.y = ra0.z = ra0.w = 0u; ra1 = ra0;                          \
    }                                                                         \
  }

#define CONV_STAGEA(BUF)                                                      \
  {                                                                           \
    uint4* pa_ = (uint4*)&As[(BUF)][lrow][lhalf * 16];                        \
    pa_[0] = ra0; pa_[1] = ra1;                                               \
  }

#if !USE_TDM
#define CONV_LOADB(STEP)                                                      \
  {                                                                           \
    int r_  = (STEP) >> 4;                                                    \
    int c0_ = ((STEP) & 15) * 32 + lhalf * 16;                                \
    const uint4* q_ = (const uint4*)(Wb +                                     \
        ((size_t)nrow * KTOT + r_ * 512 + c0_));                              \
    rb0 = q_[0]; rb1 = q_[1];                                                 \
  }
#define CONV_STAGEB(BUF)                                                      \
  {                                                                           \
    uint4* pb_ = (uint4*)&Bs[(BUF)][lrow][lhalf * 16];                        \
    pb_[0] = rb0; pb_[1] = rb1;                                               \
  }
#endif

#if USE_TDM
  const unsigned ldsB0 = (unsigned)(unsigned long long)(lds_us_t*)&Bs[0][0][0];
  const unsigned ldsB1 = ldsB0 + 128u * 40u * 2u;
#define CONV_TDM_B(STEP, BUF)                                                 \
  {                                                                           \
    int r_  = (STEP) >> 4;                                                    \
    int c0_ = ((STEP) & 15) * 32;                                             \
    tdm_load_B(Wb + ((size_t)n0 * KTOT + r_ * 512 + c0_),                     \
               (BUF) ? ldsB1 : ldsB0);                                        \
  }
#endif

  /* ---- prologue: tile 0 ---- */
#if USE_TDM
  if (wave == 0) CONV_TDM_B(0, 0)
  CONV_LOADA(0)
  CONV_STAGEA(0)
  if (wave == 0) wait_tensorcnt0();
#else
  CONV_LOADA(0)
  CONV_LOADB(0)
  CONV_STAGEA(0)
  CONV_STAGEB(0)
#endif
  __syncthreads();

  const int arow_c = wave * 16 + (lane & 15);
  const int kb     = lane >> 4;

  for (int s = 0; s < 144; ++s) {
    const int buf = s & 1;
    if (s + 1 < 144) {
#if USE_TDM
      if (wave == 0) CONV_TDM_B(s + 1, buf ^ 1)
      CONV_LOADA(s + 1)
#else
      CONV_LOADA(s + 1)
      CONV_LOADB(s + 1)
#endif
    }

    FragBF fa;
    const uint4* ar = (const uint4*)&As[buf][arow_c][0];
    fa.q[0] = ar[kb];        /* K: kb*8 .. kb*8+8   */
    fa.q[1] = ar[kb + 2];    /* K: kb*8+16 .. +24   */
#pragma unroll
    for (int nt = 0; nt < 8; ++nt) {
      FragBF fb;
      const uint4* br = (const uint4*)&Bs[buf][nt * 16 + (lane & 15)][0];
      fb.q[0] = br[kb];
      fb.q[1] = br[kb + 2];
      acc[nt] = __builtin_amdgcn_wmma_f32_16x16x32_bf16(
          false, fa.v, false, fb.v, (short)0, acc[nt], false, false);
    }

    if (s + 1 < 144) {
#if USE_TDM
      CONV_STAGEA(buf ^ 1)
      if (wave == 0) wait_tensorcnt0();
#else
      CONV_STAGEA(buf ^ 1)
      CONV_STAGEB(buf ^ 1)
#endif
    }
    __syncthreads();
  }

  /* bias + ReLU + store bf16 NHWC */
#pragma unroll
  for (int nt = 0; nt < 8; ++nt) {
    int oc = n0 + nt * 16 + (lane & 15);
    float bias = b1[oc];
    int mbase = m0 + wave * 16 + (lane >> 4) * 8;
#pragma unroll
    for (int v = 0; v < 8; ++v) {
      float val = acc[nt][v] + bias;
      val = val > 0.f ? val : 0.f;
      hb[(size_t)(mbase + v) * 512 + oc] = f2bf(val);
    }
  }
#undef CONV_LOADA
#undef CONV_STAGEA
}

/* ---------------- 1x1 heads (18 score + 36 loc) as WMMA GEMM ---------------- */

extern "C" __global__ __launch_bounds__(256) void k_head_wmma(
    const unsigned short* __restrict__ hb,   /* [16384][512] bf16 */
    const unsigned short* __restrict__ Wh,   /* [64][512]   bf16  */
    const float* __restrict__ bs, const float* __restrict__ bl,
    float* __restrict__ out_locs, float* __restrict__ out_scores)
{
  __shared__ unsigned short As[2][128][40];
  __shared__ unsigned short Bs[2][64][40];

  const int tid  = threadIdx.x;
  const int lane = tid & 31;
  const int wave = tid >> 5;
  const int m0   = blockIdx.x * 128;

  const int lrow  = tid >> 1;
  const int lhalf = tid & 1;
  const int mg    = m0 + lrow;
  const bool doB  = tid < 128;

  v8f acc[4];
  v8f vz = {};
#pragma unroll
  for (int t = 0; t < 4; ++t) acc[t] = vz;

  uint4 ra0, ra1, rb0, rb1;
  rb0.x = rb0.y = rb0.z = rb0.w = 0u; rb1 = rb0;

#define HEAD_LOAD(STEP)                                                       \
  {                                                                           \
    int c0_ = (STEP) * 32 + lhalf * 16;                                       \
    const uint4* p_ = (const uint4*)(hb + ((size_t)mg * 512 + c0_));          \
    ra0 = p_[0]; ra1 = p_[1];                                                 \
    if (doB) {                                                                \
      const uint4* q_ = (const uint4*)(Wh + ((size_t)lrow * 512 + c0_));      \
      rb0 = q_[0]; rb1 = q_[1];                                               \
    }                                                                         \
  }

#define HEAD_STAGE(BUF)                                                       \
  {                                                                           \
    uint4* pa_ = (uint4*)&As[(BUF)][lrow][lhalf * 16];                        \
    pa_[0] = ra0; pa_[1] = ra1;                                               \
    if (doB) {                                                                \
      uint4* pb_ = (uint4*)&Bs[(BUF)][lrow][lhalf * 16];                      \
      pb_[0] = rb0; pb_[1] = rb1;                                             \
    }                                                                         \
  }

  HEAD_LOAD(0)
  HEAD_STAGE(0)
  __syncthreads();

  const int arow_c = wave * 16 + (lane & 15);
  const int kb     = lane >> 4;

  for (int s = 0; s < 16; ++s) {
    const int buf = s & 1;
    if (s + 1 < 16) HEAD_LOAD(s + 1)

    FragBF fa;
    const uint4* ar = (const uint4*)&As[buf][arow_c][0];
    fa.q[0] = ar[kb];
    fa.q[1] = ar[kb + 2];
#pragma unroll
    for (int nt = 0; nt < 4; ++nt) {
      FragBF fb;
      const uint4* br = (const uint4*)&Bs[buf][nt * 16 + (lane & 15)][0];
      fb.q[0] = br[kb];
      fb.q[1] = br[kb + 2];
      acc[nt] = __builtin_amdgcn_wmma_f32_16x16x32_bf16(
          false, fa.v, false, fb.v, (short)0, acc[nt], false, false);
    }

    if (s + 1 < 16) HEAD_STAGE(buf ^ 1)
    __syncthreads();
  }

#pragma unroll
  for (int nt = 0; nt < 4; ++nt) {
    int ch = nt * 16 + (lane & 15);
    float bias = (ch < 18) ? bs[ch] : (ch < 54 ? bl[ch - 18] : 0.f);
#pragma unroll
    for (int v = 0; v < 8; ++v) {
      int m = m0 + wave * 16 + (lane >> 4) * 8 + v;
      int n = m >> 12, pos = m & 4095;
      float val = acc[nt][v] + bias;
      if (ch < 18)
        out_scores[(size_t)n * (NA_TOT * 2) + pos * 18 + ch] = val;
      else if (ch < 54)
        out_locs[(size_t)n * (NA_TOT * 4) + pos * 36 + (ch - 18)] = val;
    }
  }
#undef HEAD_LOAD
#undef HEAD_STAGE
}

/* ---------------- anchors + softmax + loc2bbox + clip + filter ---------------- */

extern "C" __global__ void k_decode(
    const float* __restrict__ locs, const float* __restrict__ scores,
    const int* __restrict__ ih_p, const int* __restrict__ iw_p,
    float* __restrict__ score_out, float* __restrict__ boxes_out,
    float* __restrict__ anch_out)
{
  int gid = blockIdx.x * blockDim.x + threadIdx.x;
  if (gid >= N_IMG * NA_TOT) return;
  int n = gid / NA_TOT, i = gid % NA_TOT;
  int pos = i / 9, a = i % 9;
  int y = pos >> 6, xx = pos & 63;

  const float ratios[3] = {0.5f, 1.0f, 2.0f};
  const float scales[3] = {8.0f, 16.0f, 32.0f};
  float rt  = ratios[a / 3];
  float scl = scales[a % 3];
  float hA = 16.f * scl * sqrtf(rt);
  float wA = 16.f * scl * sqrtf(1.f / rt);
  float cy0 = y * 16.f + 8.f, cx0 = xx * 16.f + 8.f;
  float ay1 = cy0 - 0.5f * hA, ax1 = cx0 - 0.5f * wA;
  float ay2 = cy0 + 0.5f * hA, ax2 = cx0 + 0.5f * wA;
  if (n == 0) {
    anch_out[i * 4 + 0] = ay1; anch_out[i * 4 + 1] = ax1;
    anch_out[i * 4 + 2] = ay2; anch_out[i * 4 + 3] = ax2;
  }

  const float* lp = locs + (size_t)n * NA_TOT * 4 + (size_t)i * 4;
  float dy = lp[0], dx = lp[1], dh = lp[2], dw = lp[3];
  float ah = ay2 - ay1, aw = ax2 - ax1;
  float ccy = ay1 + 0.5f * ah + dy * ah;
  float ccx = ax1 + 0.5f * aw + dx * aw;
  float h = expf(dh) * ah, w = expf(dw) * aw;

  float imh = (float)(*ih_p), imw = (float)(*iw_p);
  float y1 = fminf(fmaxf(ccy - 0.5f * h, 0.f), imh);
  float x1 = fminf(fmaxf(ccx - 0.5f * w, 0.f), imw);
  float y2 = fminf(fmaxf(ccy + 0.5f * h, 0.f), imh);
  float x2 = fminf(fmaxf(ccx + 0.5f * w, 0.f), imw);
  bool valid = (y2 - y1 >= 16.f) && (x2 - x1 >= 16.f);

  float s0 = scores[(size_t)n * NA_TOT * 2 + (size_t)i * 2 + 0];
  float s1 = scores[(size_t)n * NA_TOT * 2 + (size_t)i * 2 + 1];
  float fg = 1.f / (1.f + expf(s0 - s1));

  score_out[gid] = valid ? fg : -__builtin_inff();
  float* bo = boxes_out + (size_t)gid * 4;
  bo[0] = y1; bo[1] = x1; bo[2] = y2; bo[3] = x2;
}

extern "C" __global__ void k_hist(const float* __restrict__ score,
                                  int* __restrict__ hist) {
  int gid = blockIdx.x * blockDim.x + threadIdx.x;
  if (gid >= N_IMG * NA_TOT) return;
  float s = score[gid];
  if (s >= 0.f) {
    int b = (int)(s * 256.f); b = b > 255 ? 255 : b;
    atomicAdd(&hist[(gid / NA_TOT) * 256 + b], 1);
  }
}

/* ---------------- per-image top-3000 via histogram + LDS bitonic sort ------- */

extern "C" __global__ __launch_bounds__(1024) void k_select(
    const float* __restrict__ score, const float* __restrict__ boxes,
    const int* __restrict__ hist,
    float* __restrict__ topb, float* __restrict__ tops)
{
  extern __shared__ __align__(16) char smem[];
  float* ss = (float*)smem;                  /* CAND_MAX */
  int*   si = (int*)(ss + CAND_MAX);         /* CAND_MAX */
  int*   sh = si + CAND_MAX;                 /* 256      */
  __shared__ int cnt;
  __shared__ float thr;

  int n = blockIdx.x, tid = threadIdx.x;
  for (int b = tid; b < 256; b += 1024) sh[b] = hist[n * 256 + b];
  if (tid == 0) cnt = 0;
  __syncthreads();
  if (tid == 0) {
    int cum = 0, t = 0;
    for (int b = 255; b >= 0; --b) { cum += sh[b]; if (cum >= PRE_NMS) { t = b; break; } }
    thr = t * (1.f / 256.f);
  }
  __syncthreads();
  float th = thr;
  for (int j = tid; j < NA_TOT; j += 1024) {
    float s = score[(size_t)n * NA_TOT + j];
    if (s >= th) {
      int p = atomicAdd(&cnt, 1);
      if (p < CAND_MAX) { ss[p] = s; si[p] = j; }
    }
  }
  __syncthreads();
  int c = cnt < CAND_MAX ? cnt : CAND_MAX;
  for (int j = tid; j < CAND_MAX; j += 1024)
    if (j >= c) { ss[j] = -__builtin_inff(); si[j] = -1; }
  __syncthreads();

  /* bitonic sort, descending by score */
  for (unsigned k = 2; k <= CAND_MAX; k <<= 1) {
    for (unsigned j = k >> 1; j > 0; j >>= 1) {
      for (unsigned i = tid; i < CAND_MAX; i += 1024) {
        unsigned l = i ^ j;
        if (l > i) {
          bool desc = ((i & k) == 0);
          float a = ss[i], b2 = ss[l];
          bool sw = desc ? (a < b2) : (a > b2);
          if (sw) { ss[i] = b2; ss[l] = a; int t2 = si[i]; si[i] = si[l]; si[l] = t2; }
        }
      }
      __syncthreads();
    }
  }

  for (int t = tid; t < PRE_NMS; t += 1024) {
    float s = ss[t]; int idx = si[t];
    tops[(size_t)n * PRE_NMS + t] = s;
    float* dst = topb + ((size_t)n * PRE_NMS + t) * 4;
    if (idx >= 0 && s > -1e30f) {
      const float* src = boxes + ((size_t)n * NA_TOT + idx) * 4;
      dst[0] = src[0]; dst[1] = src[1]; dst[2] = src[2]; dst[3] = src[3];
    } else {
      dst[0] = dst[1] = dst[2] = dst[3] = 0.f;
    }
  }
}

/* ---------------- greedy NMS + ROI writeout ---------------- */

__device__ __forceinline__ float iou_fn(const float* a, const float* b) {
  float areaA = (a[2] - a[0]) * (a[3] - a[1]);
  float areaB = (b[2] - b[0]) * (b[3] - b[1]);
  float y1 = fmaxf(a[0], b[0]), x1 = fmaxf(a[1], b[1]);
  float y2 = fminf(a[2], b[2]), x2 = fminf(a[3], b[3]);
  float inter = fmaxf(y2 - y1, 0.f) * fmaxf(x2 - x1, 0.f);
  float uni = areaA + areaB - inter;
  return uni > 0.f ? inter / uni : 0.f;
}

extern "C" __global__ __launch_bounds__(1024) void k_nms(
    const float* __restrict__ topb, const float* __restrict__ tops,
    float* __restrict__ rois, float* __restrict__ valid)
{
  extern __shared__ __align__(16) char smem[];
  float* bx = (float*)smem;                       /* PRE_NMS*4 */
  float* sc = bx + PRE_NMS * 4;                   /* PRE_NMS   */
  unsigned char* keep = (unsigned char*)(sc + PRE_NMS);
  __shared__ int flag;

  int n = blockIdx.x, tid = threadIdx.x;
  for (int i = tid; i < PRE_NMS * 4; i += 1024) bx[i] = topb[(size_t)n * PRE_NMS * 4 + i];
  for (int i = tid; i < PRE_NMS; i += 1024) { sc[i] = tops[(size_t)n * PRE_NMS + i]; keep[i] = 0; }
  if (tid == 0) flag = 0;
  __syncthreads();

  for (int i = 0; i < PRE_NMS; ++i) {
    for (int j = tid; j < i; j += 1024)
      if (keep[j] && iou_fn(&bx[i * 4], &bx[j * 4]) > 0.7f) atomicOr(&flag, 1);
    __syncthreads();
    if (tid == 0) { keep[i] = (flag == 0) && (sc[i] > -1e30f); flag = 0; }
    __syncthreads();
  }

  if (tid == 0) {
    int k = 0;
    for (int i = 0; i < PRE_NMS && k < POST_NMS; ++i) {
      if (keep[i]) {
        float* r = rois + ((size_t)n * POST_NMS + k) * 4;
        r[0] = bx[i * 4 + 0]; r[1] = bx[i * 4 + 1];
        r[2] = bx[i * 4 + 2]; r[3] = bx[i * 4 + 3];
        valid[(size_t)n * POST_NMS + k] = 1.f;
        ++k;
      }
    }
    for (; k < POST_NMS; ++k) {
      float* r = rois + ((size_t)n * POST_NMS + k) * 4;
      r[0] = r[1] = r[2] = r[3] = 0.f;
      valid[(size_t)n * POST_NMS + k] = 0.f;
    }
  }
}

/* ---------------- host launcher ---------------- */

extern "C" void kernel_launch(void* const* d_in, const int* in_sizes, int n_in,
                              void* d_out, int out_size, void* d_ws, size_t ws_size,
                              hipStream_t stream) {
  const float* x  = (const float*)d_in[0];
  const float* W1 = (const float*)d_in[1];
  const float* b1 = (const float*)d_in[2];
  const float* Ws = (const float*)d_in[3];
  const float* bs = (const float*)d_in[4];
  const float* Wl = (const float*)d_in[5];
  const float* bl = (const float*)d_in[6];
  const int*   ih = (const int*)d_in[7];
  const int*   iw = (const int*)d_in[8];

  float* out        = (float*)d_out;
  float* out_locs   = out + OFF_LOCS;
  float* out_scores = out + OFF_SCORES;
  float* out_rois   = out + OFF_ROIS;
  float* out_valid  = out + OFF_VALID;
  float* out_anch   = out + OFF_ANCH;

  char* ws = (char*)d_ws;
  unsigned short* xb   = (unsigned short*)(ws + WS_XB);
  unsigned short* Wb   = (unsigned short*)(ws + WS_WB);
  unsigned short* Wh   = (unsigned short*)(ws + WS_WH);
  unsigned short* hb   = (unsigned short*)(ws + WS_HB);
  float*          fsc  = (float*)(ws + WS_SC);
  float*          bxs  = (float*)(ws + WS_BX);
  int*            hist = (int*)(ws + WS_HIST);
  float*          topb = (float*)(ws + WS_TOPB);
  float*          tops = (float*)(ws + WS_TOPS);

  k_zero_hist<<<1, 1024, 0, stream>>>(hist);
  k_cvt_x   <<<(N_IMG * 512 * NPOS + 255) / 256, 256, 0, stream>>>(x, xb);
  k_cvt_w   <<<(512 * KTOT + 255) / 256, 256, 0, stream>>>(W1, Wb);
  k_cvt_hw  <<<(64 * 512 + 255) / 256, 256, 0, stream>>>(Ws, Wl, Wh);

  k_conv_wmma<<<dim3(MTOT / 128, 512 / 128), 256, 0, stream>>>(xb, Wb, b1, hb);
  k_head_wmma<<<MTOT / 128, 256, 0, stream>>>(hb, Wh, bs, bl, out_locs, out_scores);

  k_decode<<<(N_IMG * NA_TOT + 255) / 256, 256, 0, stream>>>(
      out_locs, out_scores, ih, iw, fsc, bxs, out_anch);
  k_hist<<<(N_IMG * NA_TOT + 255) / 256, 256, 0, stream>>>(fsc, hist);

  size_t sel_sh = (size_t)CAND_MAX * 8 + 256 * 4;                 /* 33792 B */
  k_select<<<N_IMG, 1024, sel_sh, stream>>>(fsc, bxs, hist, topb, tops);

  size_t nms_sh = (size_t)PRE_NMS * 4 * 4 + PRE_NMS * 4 + PRE_NMS + 64; /* ~63 KB */
  k_nms<<<N_IMG, 1024, nms_sh, stream>>>(topb, tops, out_rois, out_valid);
}
